// GCN_33938831573040
// MI455X (gfx1250) — matrix operand order repaired and checked by reading, not compiled, not run on hardware
//
#include <hip/hip_runtime.h>
#include <hip/hip_bf16.h>

// ---------------------------------------------------------------------------
// 2-layer GCN for MI455X (gfx1250, wave32, WMMA).
//   out = GCNConv(relu(GCNConv(x, W1, b1)), W2, b2)
// GEMMs use v_wmma_f32_16x16x32_f16 with 128x64 output blocks (A fragment
// reused across 4 N-tiles); graph aggregation uses f32 atomics.
// ---------------------------------------------------------------------------

#define N_NODES   50000
#define N_EDGES   150000
#define D_FEAT    512
#define HIDDEN    512
#define N_CLASSES 47
#define KDIM      512     // K is 512 for both GEMMs
#define NTILES    4       // 16-col N tiles per block (64 cols)

typedef __attribute__((ext_vector_type(16))) _Float16 v16h;
typedef __attribute__((ext_vector_type(2)))  _Float16 v2h;
typedef __attribute__((ext_vector_type(8)))  float    v8f;

// ---------------------------- degree / norm --------------------------------

__global__ void gcn_deg_init(float* __restrict__ deg) {
    int i = blockIdx.x * blockDim.x + threadIdx.x;
    if (i < N_NODES) deg[i] = 1.0f;              // self-loop
}

__global__ void gcn_deg_acc(const int* __restrict__ cols, float* __restrict__ deg) {
    int e = blockIdx.x * blockDim.x + threadIdx.x;
    if (e < N_EDGES) atomicAdd(&deg[cols[e]], 1.0f);
}

__global__ void gcn_deg_rsqrt(float* __restrict__ deg) {
    int i = blockIdx.x * blockDim.x + threadIdx.x;
    if (i < N_NODES) deg[i] = rsqrtf(deg[i]);    // deg >= 1 always
}

// ------------------------------ WMMA GEMM ----------------------------------
// C[M x Nout] = op(A)[M x 512] * B[512 x Nout]
// op(A) = relu(A + bias_row) when RELU_BIAS, else A.
// Block = 256 threads = 8 waves, computes a 128-row x 64-col output block:
//   blockIdx.x -> 64-col N block, blockIdx.y -> chunk of 8 16-row M tiles.
// The 512x64 B panel is converted to f16 once per block into LDS (64 KB),
// pre-swizzled into WMMA B-fragment order (16 contiguous halves per lane).
// Each A fragment feeds 4 WMMAs (one per 16-col tile).

template <bool RELU_BIAS>
__global__ void gcn_gemm_wmma(const float* __restrict__ A,
                              const float* __restrict__ B,
                              const float* __restrict__ bias,   // length K, may be null
                              float* __restrict__ C,
                              int M, int Nout) {
    __shared__ alignas(32) _Float16 Bs[NTILES][KDIM * 16];   // 64 KB

    const int tid = threadIdx.x;
    const int n0  = blockIdx.x * (16 * NTILES);

    // Cooperative B-panel load: convert K-pairs (packed f16 cvt) and store b32.
    // LDS layout per tile t: element (k, c) at [(((k>>5)*2 + ((k>>3)&1))*16 + c)*16 + (k&7) + ((k>>1)&8)],
    // so the fragment for K-step s, lane L is 16 contiguous halves at
    // group g = 2*s + (L>>4), col c = L&15.
    for (int idx = tid; idx < (KDIM / 2) * (16 * NTILES); idx += 256) {
        int cw  = idx & (16 * NTILES - 1);   // 0..63: column within block
        int kk  = idx / (16 * NTILES);       // K pair index
        int k   = kk * 2;
        int t   = cw >> 4;
        int c   = cw & 15;
        int col = n0 + cw;
        float f0 = (col < Nout) ? B[(size_t)k * Nout + col]       : 0.0f;
        float f1 = (col < Nout) ? B[(size_t)(k + 1) * Nout + col] : 0.0f;
        v2h p; p[0] = (_Float16)f0; p[1] = (_Float16)f1;
        *(v2h*)&Bs[t][((((k >> 5) * 2 + ((k >> 3) & 1)) * 16 + c) << 4) | ((k & 7) + ((k >> 1) & 8))] = p;
    }
    __syncthreads();

    const int wave  = tid >> 5;
    const int lane  = tid & 31;
    const int hi    = lane >> 4;     // 0: lanes 0-15, 1: lanes 16-31
    const int lr    = lane & 15;
    const int mtile = blockIdx.y * 8 + wave;
    if (mtile * 16 >= M) return;     // uniform per wave (EXEC all-ones at WMMA)

    const int row = mtile * 16 + lr; // A row owned by this lane
    const float* __restrict__ arow = A + (size_t)row * KDIM;

    v8f acc[NTILES] = {};
    #pragma unroll 4
    for (int s = 0; s < KDIM / 32; ++s) {
        // A fragment: 16-bit A 16x32 layout — lane half selects K interleave:
        // lanes 0-15: K = s*32 + {0..7, 16..23}; lanes 16-31: +8.
        const int kb = s * 32 + hi * 8;
        float va[16];
        {
            float4 f0 = *(const float4*)(arow + kb);
            float4 f1 = *(const float4*)(arow + kb + 4);
            float4 f2 = *(const float4*)(arow + kb + 16);
            float4 f3 = *(const float4*)(arow + kb + 20);
            va[0]=f0.x; va[1]=f0.y; va[2]=f0.z; va[3]=f0.w;
            va[4]=f1.x; va[5]=f1.y; va[6]=f1.z; va[7]=f1.w;
            va[8]=f2.x; va[9]=f2.y; va[10]=f2.z; va[11]=f2.w;
            va[12]=f3.x; va[13]=f3.y; va[14]=f3.z; va[15]=f3.w;
        }
        if (RELU_BIAS) {
            float4 g0 = *(const float4*)(bias + kb);
            float4 g1 = *(const float4*)(bias + kb + 4);
            float4 g2 = *(const float4*)(bias + kb + 16);
            float4 g3 = *(const float4*)(bias + kb + 20);
            float vb[16] = {g0.x,g0.y,g0.z,g0.w, g1.x,g1.y,g1.z,g1.w,
                            g2.x,g2.y,g2.z,g2.w, g3.x,g3.y,g3.z,g3.w};
            #pragma unroll
            for (int j = 0; j < 16; ++j) {
                float t = va[j] + vb[j];
                va[j] = t > 0.0f ? t : 0.0f;
            }
        }
        v16h a;
        #pragma unroll
        for (int j = 0; j < 16; ++j) a[j] = (_Float16)va[j];

        const int boff = ((2 * s + hi) * 16 + lr) << 4;
        #pragma unroll
        for (int t = 0; t < NTILES; ++t) {
            const v16h b = *(const v16h*)(&Bs[t][boff]);
            acc[t] = __builtin_amdgcn_wmma_f32_16x16x32_f16(
                /*neg_a=*/false, a, /*neg_b=*/false, b,
                /*c_mod=*/(short)0, acc[t], /*reuse_a=*/false, /*reuse_b=*/false);
        }
    }

    // D layout: VGPR v holds M = v + 8*hi, N = lr (per 32-bit C/D 16x16 table).
    #pragma unroll
    for (int t = 0; t < NTILES; ++t) {
        int c = n0 + t * 16 + lr;
        if (c < Nout) {
            #pragma unroll
            for (int v = 0; v < 8; ++v) {
                int r = mtile * 16 + v + 8 * hi;
                C[(size_t)r * Nout + c] = acc[t][v];
            }
        }
    }
}

// -------------------------- aggregation kernels ----------------------------

// out[i,f] = (bias ? bias[f] : 0) + dinv[i]^2 * h[i,f]     (self-loop message)
__global__ void gcn_self_init(const float* __restrict__ h,
                              const float* __restrict__ dinv,
                              const float* __restrict__ bias,
                              float* __restrict__ out, int nfeat) {
    size_t idx = (size_t)blockIdx.x * blockDim.x + threadIdx.x;
    size_t total = (size_t)N_NODES * nfeat;
    if (idx >= total) return;
    int node = (int)(idx / nfeat);
    int f    = (int)(idx % nfeat);
    float d  = dinv[node];
    float v  = d * d * h[idx];
    if (bias) v += bias[f];
    out[idx] = v;
}

// out[col] += dinv[row]*dinv[col] * h[row]   (one block per edge)
__global__ void gcn_scatter(const float* __restrict__ h,
                            const float* __restrict__ dinv,
                            const int* __restrict__ rows,
                            const int* __restrict__ cols,
                            float* __restrict__ out, int nfeat) {
    int e = blockIdx.x;
    int r = rows[e];
    int c = cols[e];
    float norm = dinv[r] * dinv[c];
    const float* __restrict__ src = h   + (size_t)r * nfeat;
    float* __restrict__       dst = out + (size_t)c * nfeat;
    for (int f = threadIdx.x; f < nfeat; f += blockDim.x)
        atomicAdd(&dst[f], norm * src[f]);
}

// ------------------------------- launcher ----------------------------------

extern "C" void kernel_launch(void* const* d_in, const int* in_sizes, int n_in,
                              void* d_out, int out_size, void* d_ws, size_t ws_size,
                              hipStream_t stream) {
    const float* x  = (const float*)d_in[0];
    const int*   ei = (const int*)d_in[1];     // [2, N_EDGES]
    const float* W1 = (const float*)d_in[2];
    const float* b1 = (const float*)d_in[3];
    const float* W2 = (const float*)d_in[4];
    const float* b2 = (const float*)d_in[5];
    float* out = (float*)d_out;

    const int* rows = ei;             // edge_index[0] = source
    const int* cols = ei + N_EDGES;   // edge_index[1] = target

    // Workspace layout (floats): dinv | h1 (aliased by h2) | agg1
    float* ws   = (float*)d_ws;
    float* dinv = ws;                                   // 50048 (padded)
    float* h1   = ws + 50048;                           // 50000*512
    float* agg1 = h1 + (size_t)N_NODES * HIDDEN;        // 50000*512
    float* h2   = h1;                                   // h1 dead after scatter1

    // 1) symmetric normalization dinv = rsqrt(in_deg + 1)
    gcn_deg_init <<<(N_NODES + 255) / 256, 256, 0, stream>>>(dinv);
    gcn_deg_acc  <<<(N_EDGES + 255) / 256, 256, 0, stream>>>(cols, dinv);
    gcn_deg_rsqrt<<<(N_NODES + 255) / 256, 256, 0, stream>>>(dinv);

    // 2) h1 = x @ W1   (WMMA f16, f32 accumulate)
    {
        dim3 grid(HIDDEN / (16 * NTILES), (N_NODES / 16 + 7) / 8);
        gcn_gemm_wmma<false><<<grid, 256, 0, stream>>>(x, W1, nullptr, h1,
                                                       N_NODES, HIDDEN);
    }

    // 3) agg1 = dinv^2*h1 + scatter(norm*h1)   (b1+relu fused into GEMM2 A-load)
    {
        size_t total = (size_t)N_NODES * HIDDEN;
        gcn_self_init<<<(unsigned)((total + 255) / 256), 256, 0, stream>>>(
            h1, dinv, nullptr, agg1, HIDDEN);
        gcn_scatter<<<N_EDGES, 256, 0, stream>>>(h1, dinv, rows, cols, agg1, HIDDEN);
    }

    // 4) h2 = relu(agg1 + b1) @ W2   (single 64-col N block, cols masked to 47)
    {
        dim3 grid((N_CLASSES + 16 * NTILES - 1) / (16 * NTILES),
                  (N_NODES / 16 + 7) / 8);
        gcn_gemm_wmma<true><<<grid, 256, 0, stream>>>(agg1, W2, b1, h2,
                                                      N_NODES, N_CLASSES);
    }

    // 5) out = b2 + dinv^2*h2 + scatter(norm*h2)
    {
        size_t total = (size_t)N_NODES * N_CLASSES;
        gcn_self_init<<<(unsigned)((total + 255) / 256), 256, 0, stream>>>(
            h2, dinv, b2, out, N_CLASSES);
        gcn_scatter<<<N_EDGES, 64, 0, stream>>>(h2, dinv, rows, cols, out, N_CLASSES);
    }
}
